// TRGRAMKQV_10728828306106
// MI455X (gfx1250) — matrix operand order repaired and checked by reading, not compiled, not run on hardware
//
#include <hip/hip_runtime.h>
#include <hip/hip_bf16.h>
#include <math.h>

typedef __attribute__((ext_vector_type(16))) _Float16     v16h;
typedef __attribute__((ext_vector_type(8)))  _Float16     v8h;
typedef __attribute__((ext_vector_type(8)))  float        v8f;
typedef __attribute__((ext_vector_type(4)))  unsigned int u32x4;
typedef __attribute__((ext_vector_type(8)))  int          i32x8;
typedef __attribute__((ext_vector_type(4)))  int          i32x4;

static constexpr int Bn  = 8;
static constexpr int Cn  = 256;
static constexpr int TCn = 512;
static constexpr int Nn  = 48 * 48;        // 2304, divisible by 64
static constexpr float LOG2E = 1.44269504088896340736f;

// A-fragment (16x32 f16, M=lane%16): elems [0..7]=p[0..7] (K=lh*8+e), [8..15]=p[16..23]
static __device__ __forceinline__ v16h load_a16(const _Float16* p) {
  v8h lo = *(const v8h*)(p);
  v8h hi = *(const v8h*)(p + 16);
  v16h a;
#pragma unroll
  for (int i = 0; i < 8; ++i) { a[i] = lo[i]; a[i + 8] = hi[i]; }
  return a;
}
// B-fragment (32x16 f16, N=lane%16): elems [0..7]=p[0..7] (K=lh*8+e), [8..15]=p[16..23]
static __device__ __forceinline__ v16h load_b16(const _Float16* p) {
  v8h lo = *(const v8h*)(p);
  v8h hi = *(const v8h*)(p + 16);
  v16h b;
#pragma unroll
  for (int i = 0; i < 8; ++i) { b[i] = lo[i]; b[i + 8] = hi[i]; }
  return b;
}

// ---- Tensor Data Mover: 2-D tile load, f16 elements (data_size = 2 bytes) ----
// D# layout per CDNA5 ISA ch.8. Groups 2/3 zero (<=2D tensor).
static __device__ __forceinline__ void tdm_load_2d(
    unsigned lds_off, const void* gptr,
    unsigned tensor_d0, unsigned tensor_d1, unsigned d0_stride,
    unsigned tile_d0, unsigned tile_d1) {
  const unsigned long long ga = (unsigned long long)(size_t)gptr;
  u32x4 g0;
  g0[0] = 1u;                                              // count=1, user mode
  g0[1] = lds_off;                                         // lds_addr (bytes)
  g0[2] = (unsigned)ga;                                    // global_addr[31:0]
  g0[3] = ((unsigned)(ga >> 32) & 0x01FFFFFFu) | (2u << 30);  // addr[56:32]|type=2
  i32x8 g1;
  g1[0] = (int)(1u << 16);                                 // data_size=1 (2 bytes)
  g1[1] = (int)((tensor_d0 & 0xFFFFu) << 16);              // tensor_dim0[15:0]
  g1[2] = (int)((tensor_d0 >> 16) | ((tensor_d1 & 0xFFFFu) << 16));
  g1[3] = (int)((tensor_d1 >> 16) | (tile_d0 << 16));      // tile_dim0
  g1[4] = (int)(tile_d1 & 0xFFFFu);                        // tile_dim1, tile_dim2=0
  g1[5] = (int)d0_stride;                                  // tensor_dim0_stride lo
  g1[6] = 0;
  g1[7] = 0;
  i32x4 z4 = {0, 0, 0, 0};
#if __clang_major__ >= 23
  i32x8 z8 = {0, 0, 0, 0, 0, 0, 0, 0};
  __builtin_amdgcn_tensor_load_to_lds(g0, g1, z4, z4, z8, 0);
#else
  __builtin_amdgcn_tensor_load_to_lds(g0, g1, z4, z4, 0);
#endif
}

static __device__ __forceinline__ unsigned lds_offset_of(const void* p) {
  return (unsigned)(size_t)p;   // generic LDS pointer: low 32 bits = LDS offset
}

// out[b][o][n] = sum_k W[o][k] * X[b][k][n], o in [0,256), full O per workgroup.
// OUT_T: write Out[b][n][o] (half), with optional channel-L2 normalization * scale.
// !OUT_T: write Out[b][o][n] (half), raw.
template <int KDIM, bool DO_NORM, bool OUT_T>
__global__ __launch_bounds__(256) void proj_norm_kernel(
    const float* __restrict__ Wt, const float* __restrict__ X,
    _Float16* __restrict__ Out, float scale) {
  __shared__ _Float16 ldsIn[64 * 40];   // [n_local][k] padded stride 40
  __shared__ float    normAcc[64];

  const int b     = blockIdx.x;
  const int nBase = blockIdx.y * 64;
  const int tid   = threadIdx.x;
  const int wave  = tid >> 5;           // 0..7
  const int ln    = tid & 15;
  const int lh    = (tid >> 4) & 1;

  v8f acc[2][4] = {};                   // [o-subtile][n-subtile] 16x16 f32 frags

  const float* Xb = X + (size_t)b * KDIM * Nn + nBase;

  for (int k0 = 0; k0 < KDIM; k0 += 32) {
    __syncthreads();
    {  // stage 32k x 64n tile, transposed to [n][k], f32 -> f16
      const int n   = tid & 63;
      const int kr0 = tid >> 6;         // 0..3
#pragma unroll
      for (int kk = 0; kk < 8; ++kk) {
        const int kr = kr0 * 8 + kk;    // 0..31
        ldsIn[n * 40 + kr] = (_Float16)Xb[(size_t)(k0 + kr) * Nn + n];
      }
    }
    __syncthreads();
#pragma unroll
    for (int t = 0; t < 2; ++t) {
      const int o = (wave * 2 + t) * 16 + ln;
      const float* wr = Wt + (size_t)o * KDIM + k0 + lh * 8;
      v16h afrag;
#pragma unroll
      for (int e = 0; e < 8; ++e) {
        afrag[e]     = (_Float16)wr[e];
        afrag[e + 8] = (_Float16)wr[16 + e];
      }
#pragma unroll
      for (int nt = 0; nt < 4; ++nt) {
        v16h bfrag = load_b16(&ldsIn[(nt * 16 + ln) * 40 + lh * 8]);
        acc[t][nt] = __builtin_amdgcn_wmma_f32_16x16x32_f16(
            false, afrag, false, bfrag, (short)0, acc[t][nt], false, false);
      }
    }
  }

  if (DO_NORM) {
    __syncthreads();
    if (tid < 64) normAcc[tid] = 0.f;
    __syncthreads();
#pragma unroll
    for (int t = 0; t < 2; ++t)
#pragma unroll
      for (int nt = 0; nt < 4; ++nt) {
        float s = 0.f;
#pragma unroll
        for (int r = 0; r < 8; ++r) s += acc[t][nt][r] * acc[t][nt][r];
        atomicAdd(&normAcc[nt * 16 + ln], s);   // ds_add_f32
      }
    __syncthreads();
  }

#pragma unroll
  for (int t = 0; t < 2; ++t) {
#pragma unroll
    for (int nt = 0; nt < 4; ++nt) {
      const int n  = nBase + nt * 16 + ln;
      const int o0 = (wave * 2 + t) * 16 + lh * 8;   // rows r..r+7 contiguous
      if (OUT_T) {
        float rn = scale;
        if (DO_NORM) {
          const float nr = sqrtf(normAcc[nt * 16 + ln]);
          rn = scale / fmaxf(nr, 1e-6f);
        }
        v8h h;
#pragma unroll
        for (int r = 0; r < 8; ++r) h[r] = (_Float16)(acc[t][nt][r] * rn);
        *(v8h*)&Out[((size_t)b * Nn + n) * Cn + o0] = h;
      } else {
#pragma unroll
        for (int r = 0; r < 8; ++r)
          Out[((size_t)b * Cn + o0 + r) * Nn + n] = (_Float16)acc[t][nt][r];
      }
    }
  }
}

// Flash attention with TDM double-buffered K/V staging.
// Workgroup = 64 query rows of one batch, 4 waves; wave owns 16 rows x 256 ch.
// qh,kh: [B][N][C] half (q pre-scaled by 1/(16*||q||), k by 1/||k||).
// vh: [B][C][N] half. Fused epilogue: out = x + alpha * O.
__global__ __launch_bounds__(128) void flash_attn_kernel(
    const _Float16* __restrict__ qh, const _Float16* __restrict__ kh,
    const _Float16* __restrict__ vh, const float* __restrict__ X,
    const float* __restrict__ alphap, float* __restrict__ Out) {
  extern __shared__ __align__(16) _Float16 smem[];
  // [0)            2*64*256  : K tiles (double buffer), row-major [m_local][c]
  // [2*64*256)     2*256*64  : V tiles (double buffer), row-major [c][m_local]
  // [+2*256*64)    4*16*80   : per-wave P tiles (stride 80)
  _Float16* Kbuf = smem;
  _Float16* Vbuf = smem + 2 * 64 * Cn;
  const int b       = blockIdx.x;
  const int rowBase = blockIdx.y * 64;
  const int tid  = threadIdx.x;
  const int wave = tid >> 5;
  const int ln   = tid & 15;
  const int lh   = (tid >> 4) & 1;
  _Float16* Pw = Vbuf + 2 * Cn * 64 + wave * (16 * 80);

  const int nRow = rowBase + wave * 16;
  const _Float16* qb = qh + ((size_t)b * Nn + nRow) * Cn;
  const _Float16* kb = kh + (size_t)b * Nn * Cn;
  const _Float16* vb = vh + (size_t)b * Cn * Nn;

  v8f oacc[16] = {};            // 16 rows x 256 channels
  float mrow[8], lrow[8];
#pragma unroll
  for (int r = 0; r < 8; ++r) { mrow[r] = -1e30f; lrow[r] = 0.f; }

  constexpr int NB = Nn / 64;   // 36 column blocks

  if (wave == 0) {              // prologue: DMA block 0 into buffer parity 0
    tdm_load_2d(lds_offset_of(Kbuf), kb, Cn, Nn, Cn, Cn, 64);
    tdm_load_2d(lds_offset_of(Vbuf), vb, Nn, Cn, Nn, 64, Cn);
  }

  for (int j = 0; j < NB; ++j) {
    const int m0 = j * 64;
    const _Float16* Kt = Kbuf + (j & 1) * 64 * Cn;
    const _Float16* Vt = Vbuf + (j & 1) * Cn * 64;

    if (wave == 0) {
      if (j + 1 < NB) {   // DMA next block while this one is consumed
        const int mn = m0 + 64;
        tdm_load_2d(lds_offset_of(Kbuf + ((j + 1) & 1) * 64 * Cn),
                    kb + (size_t)mn * Cn, Cn, Nn, Cn, Cn, 64);
        tdm_load_2d(lds_offset_of(Vbuf + ((j + 1) & 1) * Cn * 64),
                    vb + mn, Nn, Cn, Nn, 64, Cn);
        __builtin_amdgcn_s_wait_tensorcnt(2);   // block j's pair complete
      } else {
        __builtin_amdgcn_s_wait_tensorcnt(0);
      }
    }
    __syncthreads();            // tile j visible to all waves

    // ---- S = qn * kn^T for 16 rows x 64 cols (K = C = 256) ----
    v8f s[4] = {};
#pragma unroll
    for (int kc = 0; kc < 8; ++kc) {
      v16h afrag = load_a16(qb + (size_t)ln * Cn + kc * 32 + lh * 8);
#pragma unroll
      for (int sub = 0; sub < 4; ++sub) {
        v16h bfrag = load_b16(Kt + (sub * 16 + ln) * Cn + kc * 32 + lh * 8);
        s[sub] = __builtin_amdgcn_wmma_f32_16x16x32_f16(
            false, afrag, false, bfrag, (short)0, s[sub], false, false);
      }
    }

    // ---- online softmax over the 64 columns ----
    float cf[8];
#pragma unroll
    for (int r = 0; r < 8; ++r) {
      float mx = fmaxf(fmaxf(s[0][r], s[1][r]), fmaxf(s[2][r], s[3][r]));
      mx = fmaxf(mx, __shfl_xor(mx, 1, 32));
      mx = fmaxf(mx, __shfl_xor(mx, 2, 32));
      mx = fmaxf(mx, __shfl_xor(mx, 4, 32));
      mx = fmaxf(mx, __shfl_xor(mx, 8, 32));
      const float mnew = fmaxf(mrow[r], mx);
      cf[r]  = exp2f((mrow[r] - mnew) * LOG2E);
      mrow[r] = mnew;
    }
    float rs[8];
#pragma unroll
    for (int r = 0; r < 8; ++r) rs[r] = 0.f;
#pragma unroll
    for (int sub = 0; sub < 4; ++sub)
#pragma unroll
      for (int r = 0; r < 8; ++r) {
        const float p = exp2f((s[sub][r] - mrow[r]) * LOG2E);
        rs[r] += p;
        Pw[(r + lh * 8) * 80 + sub * 16 + ln] = (_Float16)p;  // ds_store_b16
      }
#pragma unroll
    for (int r = 0; r < 8; ++r) {
      float t = rs[r];
      t += __shfl_xor(t, 1, 32);
      t += __shfl_xor(t, 2, 32);
      t += __shfl_xor(t, 4, 32);
      t += __shfl_xor(t, 8, 32);
      lrow[r] = lrow[r] * cf[r] + t;
    }
#pragma unroll
    for (int f = 0; f < 16; ++f)
#pragma unroll
      for (int r = 0; r < 8; ++r) oacc[f][r] *= cf[r];

    asm volatile("s_wait_dscnt 0" ::: "memory");  // P stores -> P A-frag loads

    // ---- O += P * V  (K = 64 columns, split in two 32-wide halves) ----
#pragma unroll
    for (int h2 = 0; h2 < 2; ++h2) {
      v16h pfrag = load_a16(&Pw[ln * 80 + h2 * 32 + lh * 8]);  // ds_load_b128 x2
#pragma unroll
      for (int cs = 0; cs < 16; ++cs) {
        v16h bfrag = load_b16(Vt + (cs * 16 + ln) * 64 + h2 * 32 + lh * 8);
        oacc[cs] = __builtin_amdgcn_wmma_f32_16x16x32_f16(
            false, pfrag, false, bfrag, (short)0, oacc[cs], false, false);
      }
    }
    __syncthreads();            // all waves done with tile j before overwrite
  }

  // ---- epilogue: out = x + alpha * O / l ----
  const float alpha = *alphap;
  float rl[8];
#pragma unroll
  for (int r = 0; r < 8; ++r) rl[r] = 1.0f / lrow[r];
#pragma unroll
  for (int cs = 0; cs < 16; ++cs) {
    const int c = cs * 16 + ln;
    const size_t base = ((size_t)b * Cn + c) * Nn + nRow + lh * 8;
#pragma unroll
    for (int r = 0; r < 8; ++r)
      Out[base + r] = X[base + r] + alpha * oacc[cs][r] * rl[r];
  }
}

extern "C" void kernel_launch(void* const* d_in, const int* in_sizes, int n_in,
                              void* d_out, int out_size, void* d_ws, size_t ws_size,
                              hipStream_t stream) {
  const float* x     = (const float*)d_in[0];   // [B,C,H,W]
  const float* token = (const float*)d_in[1];   // [B,TC,H,W]
  const float* Wq    = (const float*)d_in[2];   // [C,C]
  const float* Wk    = (const float*)d_in[3];   // [C,TC]
  const float* Wv    = (const float*)d_in[4];   // [C,TC]
  const float* alpha = (const float*)d_in[5];   // [1]
  float* out = (float*)d_out;

  _Float16* qh = (_Float16*)d_ws;               // [B,N,C] normalized * 1/sqrt(C)
  _Float16* kh = qh + (size_t)Bn * Nn * Cn;     // [B,N,C] normalized
  _Float16* vh = kh + (size_t)Bn * Nn * Cn;     // [B,C,N] raw
  (void)in_sizes; (void)n_in; (void)out_size; (void)ws_size;

  const dim3 gridA(Bn, Nn / 64);
  proj_norm_kernel<Cn,  true,  true ><<<gridA, 256, 0, stream>>>(Wq, x,     qh, 1.0f / 16.0f);
  proj_norm_kernel<TCn, true,  true ><<<gridA, 256, 0, stream>>>(Wk, token, kh, 1.0f);
  proj_norm_kernel<TCn, false, false><<<gridA, 256, 0, stream>>>(Wv, token, vh, 1.0f);

  // dynamic LDS: 2x(64x256) K + 2x(256x64) V + 4x(16x80) P tiles, all f16
  const size_t smemBytes =
      (size_t)(2 * 64 * Cn + 2 * Cn * 64 + 4 * 16 * 80) * sizeof(_Float16);
  const dim3 gridB(Bn, Nn / 64);
  flash_attn_kernel<<<gridB, 128, smemBytes, stream>>>(qh, kh, vh, x, alpha, out);
}
